// MAVAP_52639119180116
// MI455X (gfx1250) — compile-verified, matrix operand order checked by reading, not run on hardware
//
#include <hip/hip_runtime.h>
#include <math.h>

// ---------------------------------------------------------------------------
// Problem constants (match reference)
// ---------------------------------------------------------------------------
#define BB   256   // batch
#define TT   128   // timesteps
#define AACT 64
#define HH   512
#define ZZ   128
#define NHH  8

typedef __attribute__((ext_vector_type(16))) _Float16 v16h;
typedef __attribute__((ext_vector_type(8)))  float    v8f;
typedef __attribute__((ext_vector_type(8)))  _Float16 h8;

__device__ __forceinline__ float sp_softplus(float x) {
    return x > 20.f ? x : log1pf(expf(x));
}
__device__ __forceinline__ float sp_relu(float x) { return x > 0.f ? x : 0.f; }
__device__ __forceinline__ float sp_sigmoid(float x) { return 1.f / (1.f + expf(-x)); }

// ---------------------------------------------------------------------------
// fp32 -> f16 weight conversion (run once per launch; weights then live in L2
// at half width for the whole T-loop)
// ---------------------------------------------------------------------------
__global__ void k_cvt_f16(const float* __restrict__ src, _Float16* __restrict__ dst, int n) {
    int i = blockIdx.x * 256 + threadIdx.x;
    if (i < n) dst[i] = (_Float16)src[i];
}

__global__ void k_zero(float* __restrict__ p, int n) {
    int i = blockIdx.x * 256 + threadIdx.x;
    if (i < n) p[i] = 0.f;
}

// ---------------------------------------------------------------------------
// Generic batched NT GEMM:  C = act(A[M,K] @ Bw[N,K]^T + bias)
//   A fp32 (converted to f16 while staging to LDS), Bw f16, C fp32.
//   Tile: 128x128 per workgroup, 8 waves (wave32), wave tile 64x32,
//   K staged 32 at a time => v_wmma_f32_16x16x32_f16, 8 per wave per K-slab.
// ---------------------------------------------------------------------------
#define Bb 128
#define Bn 128
#define Bk 32
#define LDSW 48   // row stride in halves: 96B, multiple of 16B for b128 LDS ops

__global__ __launch_bounds__(256)
void k_gemm_nt(const float* __restrict__ A, int lda, long long sA,
               const _Float16* __restrict__ Bw, int ldb, long long sB,
               float* __restrict__ C, int ldc, long long sC,
               const float* __restrict__ bias, long long sBias,
               int M, int N, int K, int act)
{
    __shared__ _Float16 As[128][LDSW] __attribute__((aligned(16)));
    __shared__ _Float16 Bs[128][LDSW] __attribute__((aligned(16)));

    const int tid  = threadIdx.x;
    const int bz   = blockIdx.z;
    A    += (size_t)bz * sA;
    Bw   += (size_t)bz * sB;
    C    += (size_t)bz * sC;
    bias += (size_t)bz * sBias;

    const int r0 = blockIdx.y * Bb;
    const int c0 = blockIdx.x * Bn;

    const int lane = tid & 31;
    const int wave = tid >> 5;
    const int wm   = wave & 1;   // 2 waves along M: 64 rows each
    const int wn   = wave >> 1;  // 4 waves along N: 32 cols each
    const int lr   = lane & 15;  // row/col within 16
    const int kg   = lane >> 4;  // K-group select (ISA 16-bit A layout)

    v8f acc[4][2] = {};

    for (int k0 = 0; k0 < K; k0 += Bk) {
        // ---- stage A tile: 128x32 fp32 -> f16 LDS -------------------------
        #pragma unroll
        for (int i = 0; i < 4; ++i) {
            int f = tid + i * 256;            // 1024 float4 slots
            int r = f >> 3;
            int c = (f & 7) << 2;
            const float4 v = *(const float4*)(A + (size_t)(r0 + r) * lda + k0 + c);
            As[r][c + 0] = (_Float16)v.x;
            As[r][c + 1] = (_Float16)v.y;
            As[r][c + 2] = (_Float16)v.z;
            As[r][c + 3] = (_Float16)v.w;
        }
        // ---- stage B tile: 128x32 f16 ------------------------------------
        #pragma unroll
        for (int i = 0; i < 2; ++i) {
            int f = tid + i * 256;            // 512 h8 slots
            int r = f >> 2;
            int c = (f & 3) << 3;
            h8 v = {};
            if (c0 + r < N)
                v = *(const h8*)(Bw + (size_t)(c0 + r) * ldb + k0 + c);
            *(h8*)&Bs[r][c] = v;
        }
        // prefetch next K-slab (global_prefetch_b8 on gfx1250)
        if (k0 + Bk < K) {
            int pr = tid >> 1;
            __builtin_prefetch(A + (size_t)(r0 + pr) * lda + k0 + Bk, 0, 3);
            if (c0 + pr < N)
                __builtin_prefetch(Bw + (size_t)(c0 + pr) * ldb + k0 + Bk, 0, 3);
        }
        __syncthreads();

        // ---- fragments (ISA 7.12.2 16-bit layouts) -----------------------
        v16h afr[4], bfr[2];
        #pragma unroll
        for (int i = 0; i < 4; ++i) {
            int r = wm * 64 + i * 16 + lr;
            union { v16h v; h8 p[2]; } u;
            u.p[0] = *(const h8*)&As[r][kg * 8];        // K = kg*8 .. +7
            u.p[1] = *(const h8*)&As[r][16 + kg * 8];   // K = 16+kg*8 .. +7
            afr[i] = u.v;
        }
        #pragma unroll
        for (int j = 0; j < 2; ++j) {
            int r = wn * 32 + j * 16 + lr;
            union { v16h v; h8 p[2]; } u;
            u.p[0] = *(const h8*)&Bs[r][kg * 8];
            u.p[1] = *(const h8*)&Bs[r][16 + kg * 8];
            bfr[j] = u.v;
        }

        #pragma unroll
        for (int i = 0; i < 4; ++i)
            #pragma unroll
            for (int j = 0; j < 2; ++j)
                acc[i][j] = __builtin_amdgcn_wmma_f32_16x16x32_f16(
                    false, afr[i], false, bfr[j], (short)0, acc[i][j], false, false);

        __syncthreads();
    }

    // ---- epilogue: bias + activation, fp32 store --------------------------
    #pragma unroll
    for (int i = 0; i < 4; ++i) {
        #pragma unroll
        for (int j = 0; j < 2; ++j) {
            int col = c0 + wn * 32 + j * 16 + lr;
            if (col >= N) continue;
            float bv = bias[col];
            int rbase = r0 + wm * 64 + i * 16 + kg * 8;   // C/D layout: M = r + 8*lane[4]
            #pragma unroll
            for (int r = 0; r < 8; ++r) {
                float v = acc[i][j][r] + bv;
                if (act) v = sp_relu(v);
                C[(size_t)(rbase + r) * ldc + col] = v;
            }
        }
    }
}

// ---------------------------------------------------------------------------
// Elementwise helpers
// ---------------------------------------------------------------------------

// phi_d: x1[b, 512+j] = relu(durs[b,t]*W_dur[j] + b_dur[j])
__global__ void k_phid(const float* __restrict__ durs, int t,
                       const float* __restrict__ W_dur, const float* __restrict__ b_dur,
                       float* __restrict__ x1) {
    int i = blockIdx.x * 256 + threadIdx.x;     // 256*512
    int b = i >> 9, j = i & 511;
    float d = durs[b * TT + t];
    x1[b * 1024 + 512 + j] = sp_relu(d * W_dur[j] + b_dur[j]);
}

// dst[256,1024] = concat(s0[256,512], s1[256,512])
__global__ void k_concat(float* __restrict__ dst,
                         const float* __restrict__ s0, const float* __restrict__ s1) {
    int i = blockIdx.x * 256 + threadIdx.x;     // 256*1024
    int b = i >> 10, j = i & 1023;
    dst[i] = (j < 512) ? s0[b * 512 + j] : s1[b * 512 + (j - 512)];
}

// latent reparameterization + per-step output slices
__global__ void k_latents(const float* __restrict__ ph, const float* __restrict__ rh,
                          const float* __restrict__ gpost, const float* __restrict__ gprior,
                          const float* __restrict__ eps, float* __restrict__ z,
                          float* __restrict__ o_pomu, float* __restrict__ o_posig,
                          float* __restrict__ o_prmu, float* __restrict__ o_prsig) {
    int i = blockIdx.x * 256 + threadIdx.x;     // 256*128
    int b = i >> 7, zi = i & 127;
    float gp = 0.f, gr = 0.f;
    #pragma unroll
    for (int n = 0; n < NHH; ++n) {
        gp += gpost[n]  * ph[b * 1024 + n * 128 + zi];
        gr += gprior[n] * rh[b * 1024 + n * 128 + zi];
    }
    float pmu = sp_relu(gp), psig = sp_softplus(gp);
    float rmu = sp_relu(gr), rsig = sp_softplus(gr);
    z[i] = eps[i] * psig + pmu;
    o_pomu[i] = pmu; o_posig[i] = psig;
    o_prmu[i] = rmu; o_prsig[i] = rsig;
}

// final prior sample: z_f = eps*softplus(gr) + relu(gr)
__global__ void k_final_latent(const float* __restrict__ rh, const float* __restrict__ gprior,
                               const float* __restrict__ eps, float* __restrict__ z) {
    int i = blockIdx.x * 256 + threadIdx.x;     // 256*128
    int b = i >> 7, zi = i & 127;
    float gr = 0.f;
    #pragma unroll
    for (int n = 0; n < NHH; ++n) gr += gprior[n] * rh[b * 1024 + n * 128 + zi];
    z[i] = eps[i] * sp_softplus(gr) + sp_relu(gr);
}

// GRU gates, in-place hidden update
__global__ void k_gate(const float* __restrict__ gi, const float* __restrict__ gh,
                       float* __restrict__ h) {
    int i = blockIdx.x * 256 + threadIdx.x;     // 256*512
    int b = i >> 9, j = i & 511;
    const float* gib = gi + (size_t)b * 1536;
    const float* ghb = gh + (size_t)b * 1536;
    float r = sp_sigmoid(gib[j]        + ghb[j]);
    float u = sp_sigmoid(gib[512 + j]  + ghb[512 + j]);
    float n = tanhf(gib[1024 + j] + r * ghb[1024 + j]);
    float hv = h[i];
    h[i] = (1.f - u) * n + u * hv;
}

// dur_dec: per-batch dot over 1024 + bias; also softplus branch
__global__ void k_dur(const float* __restrict__ cat, const float* __restrict__ Wd,
                      const float* __restrict__ bd,
                      float* __restrict__ o_dur, float* __restrict__ o_sdur) {
    __shared__ float red[256];
    int b = blockIdx.x, tid = threadIdx.x;
    float s = 0.f;
    for (int k = tid; k < 1024; k += 256) s += cat[(size_t)b * 1024 + k] * Wd[k];
    red[tid] = s;
    __syncthreads();
    for (int off = 128; off > 0; off >>= 1) {
        if (tid < off) red[tid] += red[tid + off];
        __syncthreads();
    }
    if (tid == 0) {
        float v = red[0] + bd[0];
        o_dur[b]  = v;
        o_sdur[b] = sp_softplus(v);
    }
}

// ---------------------------------------------------------------------------
// Host orchestration
// ---------------------------------------------------------------------------
extern "C" void kernel_launch(void* const* d_in, const int* in_sizes, int n_in,
                              void* d_out, int out_size, void* d_ws, size_t ws_size,
                              hipStream_t stream) {
    (void)in_sizes; (void)n_in; (void)out_size; (void)ws_size;

    // ---- inputs (setup_inputs order) ----
    const float* acts      = (const float*)d_in[0];
    const float* durs      = (const float*)d_in[1];
    const float* W_act     = (const float*)d_in[2];
    const float* b_act     = (const float*)d_in[3];
    const float* W_dur     = (const float*)d_in[4];
    const float* b_dur     = (const float*)d_in[5];
    const float* W_x       = (const float*)d_in[6];
    const float* b_x       = (const float*)d_in[7];
    const float* W_zemb    = (const float*)d_in[8];
    const float* b_zemb    = (const float*)d_in[9];
    const float* Wp1       = (const float*)d_in[10];
    const float* bp1       = (const float*)d_in[11];
    const float* Wp2       = (const float*)d_in[12];
    const float* bp2       = (const float*)d_in[13];
    const float* Wr1       = (const float*)d_in[14];
    const float* br1       = (const float*)d_in[15];
    const float* Wr2       = (const float*)d_in[16];
    const float* br2       = (const float*)d_in[17];
    const float* gpost     = (const float*)d_in[18];
    const float* gprior    = (const float*)d_in[19];
    const float* W_dec     = (const float*)d_in[20];
    const float* b_dec     = (const float*)d_in[21];
    const float* W_out     = (const float*)d_in[22];
    const float* b_out     = (const float*)d_in[23];
    const float* W_durdec  = (const float*)d_in[24];
    const float* b_durdec  = (const float*)d_in[25];
    const float* gru_Wih   = (const float*)d_in[26];
    const float* gru_Whh   = (const float*)d_in[27];
    const float* gru_bih   = (const float*)d_in[28];
    const float* gru_bhh   = (const float*)d_in[29];
    const float* eps_seq   = (const float*)d_in[30];
    const float* eps_final = (const float*)d_in[31];

    float* out = (float*)d_out;
    const size_t SL      = (size_t)TT * BB * ZZ;   // 4194304
    const size_t o_pred  = 0;
    const size_t o_dur   = (size_t)BB * AACT;      // 16384
    const size_t o_sdur  = o_dur + BB;
    const size_t o_prmu  = o_sdur + BB;
    const size_t o_prsig = o_prmu + SL;
    const size_t o_pomu  = o_prsig + SL;
    const size_t o_posig = o_pomu + SL;

    // ---- workspace carve-out ----
    size_t off = 0;
    auto ah = [&](size_t n) {
        _Float16* p = (_Float16*)((char*)d_ws + off);
        off = (off + n * sizeof(_Float16) + 255) & ~(size_t)255;
        return p;
    };
    auto af = [&](size_t n) {
        float* p = (float*)((char*)d_ws + off);
        off = (off + n * sizeof(float) + 255) & ~(size_t)255;
        return p;
    };
    _Float16* hW_act  = ah(512 * 64);
    _Float16* hW_x    = ah(512 * 1024);
    _Float16* hW_zemb = ah(512 * 128);
    _Float16* hWp1    = ah((size_t)8 * 512 * 1024);
    _Float16* hWp2    = ah((size_t)8 * 128 * 512);
    _Float16* hWr1    = ah((size_t)8 * 512 * 512);
    _Float16* hWr2    = ah((size_t)8 * 128 * 512);
    _Float16* hW_dec  = ah(512 * 1024);
    _Float16* hW_out  = ah(64 * 512);
    _Float16* hWih    = ah((size_t)1536 * 1024);
    _Float16* hWhh    = ah((size_t)1536 * 512);

    float* x1     = af((size_t)BB * 1024);
    float* phi_x  = af((size_t)BB * 512);
    float* xh     = af((size_t)BB * 1024);
    float* xz     = af((size_t)BB * 1024);
    float* ph1    = af((size_t)BB * 4096);
    float* phb    = af((size_t)BB * 1024);
    float* rh1    = af((size_t)BB * 4096);
    float* rhb    = af((size_t)BB * 1024);
    float* zbuf   = af((size_t)BB * 128);
    float* phi_z  = af((size_t)BB * 512);
    float* gi     = af((size_t)BB * 1536);
    float* gh     = af((size_t)BB * 1536);
    float* hbuf   = af((size_t)BB * 512);
    float* dec    = af((size_t)BB * 512);
    float* phi_pa = af((size_t)BB * 512);

    auto cvt = [&](const float* s, _Float16* d, size_t n) {
        k_cvt_f16<<<dim3((unsigned)((n + 255) / 256)), dim3(256), 0, stream>>>(s, d, (int)n);
    };
    cvt(W_act, hW_act, 512 * 64);
    cvt(W_x, hW_x, 512 * 1024);
    cvt(W_zemb, hW_zemb, 512 * 128);
    cvt(Wp1, hWp1, (size_t)8 * 512 * 1024);
    cvt(Wp2, hWp2, (size_t)8 * 128 * 512);
    cvt(Wr1, hWr1, (size_t)8 * 512 * 512);
    cvt(Wr2, hWr2, (size_t)8 * 128 * 512);
    cvt(W_dec, hW_dec, 512 * 1024);
    cvt(W_out, hW_out, 64 * 512);
    cvt(gru_Wih, hWih, (size_t)1536 * 1024);
    cvt(gru_Whh, hWhh, (size_t)1536 * 512);

    k_zero<<<dim3(512), dim3(256), 0, stream>>>(hbuf, BB * 512);

    auto gemm = [&](const float* A, int lda, long long sA,
                    const _Float16* Bw, int ldb, long long sB,
                    float* C, int ldc, long long sC,
                    const float* bias, long long sBias,
                    int M, int N, int K, int act, int batch) {
        dim3 g((unsigned)((N + 127) / 128), (unsigned)((M + 127) / 128), (unsigned)batch);
        k_gemm_nt<<<g, dim3(256), 0, stream>>>(A, lda, sA, Bw, ldb, sB,
                                               C, ldc, sC, bias, sBias, M, N, K, act);
    };

    // ===================== sequential scan over T =====================
    for (int t = 0; t < TT; ++t) {
        // phi_a -> x1[:, :512]
        gemm(acts + (size_t)t * AACT, TT * AACT, 0, hW_act, AACT, 0,
             x1, 1024, 0, b_act, 0, BB, 512, AACT, 1, 1);
        // phi_d -> x1[:, 512:]
        k_phid<<<dim3(512), dim3(256), 0, stream>>>(durs, t, W_dur, b_dur, x1);
        // phi_x
        gemm(x1, 1024, 0, hW_x, 1024, 0, phi_x, 512, 0, b_x, 0, BB, 512, 1024, 1, 1);
        // xh = [phi_x, h]
        k_concat<<<dim3(1024), dim3(256), 0, stream>>>(xh, phi_x, hbuf);
        // posterior head layer 1 (all heads fused: N = 8*512)
        gemm(xh, 1024, 0, hWp1, 1024, 0, ph1, 4096, 0, bp1, 0, BB, 4096, 1024, 1, 1);
        // posterior head layer 2, batched per head (block-diagonal)
        gemm(ph1, 4096, 512, hWp2, 512, 128 * 512, phb, 1024, 128, bp2, 128,
             BB, 128, 512, 1, NHH);
        // prior head layers
        gemm(hbuf, 512, 0, hWr1, 512, 0, rh1, 4096, 0, br1, 0, BB, 4096, 512, 1, 1);
        gemm(rh1, 4096, 512, hWr2, 512, 128 * 512, rhb, 1024, 128, br2, 128,
             BB, 128, 512, 1, NHH);
        // latents + output slices
        k_latents<<<dim3(128), dim3(256), 0, stream>>>(
            phb, rhb, gpost, gprior, eps_seq + (size_t)t * BB * ZZ, zbuf,
            out + o_pomu + (size_t)t * BB * ZZ, out + o_posig + (size_t)t * BB * ZZ,
            out + o_prmu + (size_t)t * BB * ZZ, out + o_prsig + (size_t)t * BB * ZZ);
        // phi_z
        gemm(zbuf, 128, 0, hW_zemb, 128, 0, phi_z, 512, 0, b_zemb, 0, BB, 512, 128, 1, 1);
        // GRU input x = [phi_x, phi_z]
        k_concat<<<dim3(1024), dim3(256), 0, stream>>>(xz, phi_x, phi_z);
        gemm(xz, 1024, 0, hWih, 1024, 0, gi, 1536, 0, gru_bih, 0, BB, 1536, 1024, 0, 1);
        gemm(hbuf, 512, 0, hWhh, 512, 0, gh, 1536, 0, gru_bhh, 0, BB, 1536, 512, 0, 1);
        k_gate<<<dim3(512), dim3(256), 0, stream>>>(gi, gh, hbuf);
    }

    // ===================== tail: final prior + decoder =====================
    gemm(hbuf, 512, 0, hWr1, 512, 0, rh1, 4096, 0, br1, 0, BB, 4096, 512, 1, 1);
    gemm(rh1, 4096, 512, hWr2, 512, 128 * 512, rhb, 1024, 128, br2, 128,
         BB, 128, 512, 1, NHH);
    k_final_latent<<<dim3(128), dim3(256), 0, stream>>>(rhb, gprior, eps_final, zbuf);
    gemm(zbuf, 128, 0, hW_zemb, 128, 0, phi_z, 512, 0, b_zemb, 0, BB, 512, 128, 1, 1);
    k_concat<<<dim3(1024), dim3(256), 0, stream>>>(xh, phi_z, hbuf);
    gemm(xh, 1024, 0, hW_dec, 1024, 0, dec, 512, 0, b_dec, 0, BB, 512, 1024, 1, 1);
    // pred_act (no activation) straight into d_out
    gemm(dec, 512, 0, hW_out, 512, 0, out + o_pred, AACT, 0, b_out, 0,
         BB, AACT, 512, 0, 1);
    // phi_pa = relu(pred_act @ W_act^T)
    gemm(out + o_pred, AACT, 0, hW_act, AACT, 0, phi_pa, 512, 0, b_act, 0,
         BB, 512, AACT, 1, 1);
    k_concat<<<dim3(1024), dim3(256), 0, stream>>>(xz, phi_pa, dec);
    k_dur<<<dim3(256), dim3(256), 0, stream>>>(xz, W_durdec, b_durdec,
                                               out + o_dur, out + o_sdur);
}
